// Seq2Seq_43181601194051
// MI455X (gfx1250) — matrix operand order repaired and checked by reading, not compiled, not run on hardware
//
#include <hip/hip_runtime.h>
#include <hip/hip_bf16.h>

// Seq2Seq LSTM (2-layer encoder + 2-layer decoder), H=64, I=27, T=256, B=4096.
// 256 workgroups, each owns a 16-row batch tile and runs the whole sequential
// time loop with state in LDS and weight WMMA B-fragments pinned in VGPRs.
// Matmuls: v_wmma_f32_16x16x32_f16 (f16 in, f32 accumulate).
// Input x is staged in 32-step LDS chunks so no global-load wait sits on the
// per-step critical path (refill cost amortized 32x).

typedef __attribute__((ext_vector_type(16))) _Float16 v16h;
typedef __attribute__((ext_vector_type(8)))  float    v8f;

#define TSTEPS 256
#define BATCH  4096
#define HID    64
#define INF    27     // input feature size
#define XCH    32     // x chunk depth (time steps staged per refill)

// ---- fast activations -------------------------------------------------------
__device__ __forceinline__ float fast_rcp(float x) {
#if __has_builtin(__builtin_amdgcn_rcpf)
    return __builtin_amdgcn_rcpf(x);        // v_rcp_f32
#else
    return 1.0f / x;
#endif
}
__device__ __forceinline__ float fast_sigmoid(float x) {
    return fast_rcp(1.0f + __expf(-x));     // v_exp_f32 + v_rcp_f32
}
__device__ __forceinline__ float fast_tanh(float x) {
#if __has_builtin(__builtin_amdgcn_tanhf)
    return __builtin_amdgcn_tanhf(x);       // v_tanh_f32 (confirmed on gfx1250)
#else
    return 1.0f - 2.0f * fast_rcp(1.0f + __expf(2.0f * x));
#endif
}

// ---- WMMA fragment helpers (layouts per CDNA5 ISA 7.12.2, wave32) -----------

// A-matrix 16x32 f16 fragment from row-major f16 LDS array (row stride elems).
// lanes 0-15: row M=lane, K = {0..7, 16..23}; lanes 16-31: row M=lane-16, K+=8.
__device__ __forceinline__ v16h load_a_frag(const _Float16* base, int stride,
                                            int ka, int lane) {
    const int m  = lane & 15;
    const int kb = (lane >> 4) * 8;
    const _Float16* row = base + m * stride + ka + kb;
    v16h a;
#pragma unroll
    for (int v = 0; v < 8; ++v) {
        const int k = (v < 4) ? (2 * v) : (2 * v + 8);   // 0,2,4,6,16,18,20,22
        a[2 * v]     = row[k];
        a[2 * v + 1] = row[k + 1];
    }
    return a;
}

// B-matrix 32x16 f16 fragment from global f32 weights W[g][k] (row length Lk).
// Covers gate columns [g0,g0+16), K rows [k0,k0+32); zero-pad K >= Lk.
__device__ __forceinline__ v16h load_b_frag(const float* W, int Lk, int g0,
                                            int k0, int lane) {
    const int n  = lane & 15;
    const int kb = (lane >> 4) * 8;
    const float* row = W + (size_t)(g0 + n) * Lk;
    v16h b;
#pragma unroll
    for (int v = 0; v < 8; ++v) {
        const int k = k0 + kb + ((v < 4) ? (2 * v) : (2 * v + 8));
        const float x0 = (k     < Lk) ? row[k]     : 0.0f;
        const float x1 = (k + 1 < Lk) ? row[k + 1] : 0.0f;
        b[2 * v]     = (_Float16)x0;
        b[2 * v + 1] = (_Float16)x1;
    }
    return b;
}

// Scatter C/D tile (16x16 f32) into LDS gate buffer G[16][256].
__device__ __forceinline__ void store_c_tile(float* G, int ntile, int lane,
                                             v8f c) {
    const int n  = ntile * 16 + (lane & 15);
    const int mb = (lane >> 4) * 8;
#pragma unroll
    for (int v = 0; v < 8; ++v) G[(mb + v) * 256 + n] = c[v];
}

// LSTM cell elementwise update for element e (= r*64 + n).
__device__ __forceinline__ float cell_update(const float* G, float* C,
                                             _Float16* Hdst, int strideH,
                                             int e) {
    const int r = e >> 6, n = e & 63;
    const float* g = G + r * 256 + n;
    const float xi = g[0], xf = g[64], xg = g[128], xo = g[192];
    const float si = fast_sigmoid(xi);
    const float sf = fast_sigmoid(xf);
    const float so = fast_sigmoid(xo);
    const float c  = sf * C[e] + si * fast_tanh(xg);
    C[e] = c;
    const float h = so * fast_tanh(c);
    Hdst[r * strideH + n] = (_Float16)h;
    return h;
}

#define WMMA(A, B, ACC) \
    __builtin_amdgcn_wmma_f32_16x16x32_f16(false, (A), false, (B), (short)0, (ACC), false, false)

__device__ __forceinline__ v8f bias_acc(float bv) {
    v8f acc;
#pragma unroll
    for (int v = 0; v < 8; ++v) acc[v] = bv;
    return acc;
}

// ---- Encoder: 2 fused LSTM layers over T=256; emits h1[T-1] -> dec_in -------
__global__ __launch_bounds__(512) void lstm_encoder(
    const float* __restrict__ x,     // (T, B, 27)
    const float* __restrict__ Wih0,  // (256, 27)
    const float* __restrict__ Whh0,  // (256, 64)
    const float* __restrict__ b0,    // (256,)
    const float* __restrict__ Wih1,  // (256, 64)
    const float* __restrict__ Whh1,  // (256, 64)
    const float* __restrict__ b1,    // (256,)
    float* __restrict__ dec_in)      // (B, 64) out
{
    // x chunk: XCH steps x 16 rows x 32 padded features, in A-fragment layout.
    __shared__ _Float16 Xc[XCH * 16 * 32];   // 32 KB
    // h state (f16): cols [0,64)=h0, [64,128)=h1
    __shared__ _Float16 Ab[16 * 128];
    __shared__ float G[16 * 256];
    __shared__ float C0[16 * 64], C1[16 * 64];

    const int tid   = threadIdx.x;
    const int lane  = tid & 31;
    const int wv    = tid >> 5;            // wave id == gate N-tile
    const int bbase = blockIdx.x * 16;

    // Persistent weight fragments (constant across all 256 steps) in VGPRs.
    const v16h w0ih = load_b_frag(Wih0, INF, wv * 16,  0, lane);   // K 27->pad32
    const v16h w0h0 = load_b_frag(Whh0, 64,  wv * 16,  0, lane);
    const v16h w0h1 = load_b_frag(Whh0, 64,  wv * 16, 32, lane);
    const v16h w1i0 = load_b_frag(Wih1, 64,  wv * 16,  0, lane);
    const v16h w1i1 = load_b_frag(Wih1, 64,  wv * 16, 32, lane);
    const v16h w1h0 = load_b_frag(Whh1, 64,  wv * 16,  0, lane);
    const v16h w1h1 = load_b_frag(Whh1, 64,  wv * 16, 32, lane);
    const float bv0 = b0[wv * 16 + (lane & 15)];
    const float bv1 = b1[wv * 16 + (lane & 15)];

    // Refill role: thread tid -> (step-in-chunk s, row r); 32*16 == 512 threads.
    const int xs = tid >> 4;               // 0..31
    const int xr = tid & 15;               // 0..15

    for (int e = tid; e < 16 * 64; e += 512) { C0[e] = 0.0f; C1[e] = 0.0f; }
    for (int e = tid; e < 16 * 128; e += 512) Ab[e] = (_Float16)0.0f;
    __syncthreads();

    for (int t = 0; t < TSTEPS; ++t) {
        // ---- chunk refill: stage x[t .. t+31] into LDS (8 refills total) ----
        if ((t & (XCH - 1)) == 0) {
            const float* src =
                x + (size_t)(t + xs) * BATCH * INF + (size_t)(bbase + xr) * INF;
            _Float16* dst = Xc + (size_t)tid * 32;   // (xs*16 + xr) * 32
#pragma unroll
            for (int k = 0; k < INF; ++k)
                dst[k] = (_Float16)__builtin_nontemporal_load(src + k);
#pragma unroll
            for (int k = INF; k < 32; ++k) dst[k] = (_Float16)0.0f;
            __syncthreads();
        }
        const _Float16* xbase = Xc + (size_t)(t & (XCH - 1)) * 16 * 32;

        // Layer 0 gates: b0 + x_t@Wih0^T + h0@Whh0^T   (3 WMMAs / wave)
        {
            const v16h a0 = load_a_frag(xbase, 32, 0, lane);
            const v16h a1 = load_a_frag(Ab, 128,  0, lane);
            const v16h a2 = load_a_frag(Ab, 128, 32, lane);
            v8f acc = bias_acc(bv0);
            acc = WMMA(a0, w0ih, acc);
            acc = WMMA(a1, w0h0, acc);
            acc = WMMA(a2, w0h1, acc);
            store_c_tile(G, wv, lane, acc);
        }
        __syncthreads();

        cell_update(G, C0, Ab, 128, tid);              // h0 -> Ab cols [0,64)
        cell_update(G, C0, Ab, 128, tid + 512);
        __syncthreads();

        // Layer 1 gates: b1 + h0@Wih1^T + h1@Whh1^T   (4 WMMAs / wave)
        {
            const v16h a0 = load_a_frag(Ab, 128,  0, lane);
            const v16h a1 = load_a_frag(Ab, 128, 32, lane);
            const v16h a2 = load_a_frag(Ab, 128, 64, lane);
            const v16h a3 = load_a_frag(Ab, 128, 96, lane);
            v8f acc = bias_acc(bv1);
            acc = WMMA(a0, w1i0, acc);
            acc = WMMA(a1, w1i1, acc);
            acc = WMMA(a2, w1h0, acc);
            acc = WMMA(a3, w1h1, acc);
            store_c_tile(G, wv, lane, acc);
        }
        __syncthreads();

        // cell1 -> h1 (cols [64,128)); at last step emit dec_in (contiguous)
        {
            const float h_a = cell_update(G, C1, Ab + 64, 128, tid);
            const float h_b = cell_update(G, C1, Ab + 64, 128, tid + 512);
            if (t == TSTEPS - 1) {
                dec_in[(size_t)bbase * HID + tid]       = h_a;
                dec_in[(size_t)bbase * HID + tid + 512] = h_b;
            }
        }
        __syncthreads();
    }
}

// ---- Decoder: 2-layer LSTM, constant input dec_in, 256 steps ----------------
__global__ __launch_bounds__(512) void lstm_decoder(
    const float* __restrict__ dec_in,  // (B, 64)
    const float* __restrict__ W1i, const float* __restrict__ W1h,
    const float* __restrict__ bd1,
    const float* __restrict__ W2i, const float* __restrict__ W2h,
    const float* __restrict__ bd2,
    float* __restrict__ out)           // (T, B, 64)
{
    // A staging (f16): cols [0,64)=h1 (dec_in during pre1), [64,128)=h2
    __shared__ _Float16 Ab[16 * 128];
    __shared__ float G[16 * 256];
    __shared__ float C1[16 * 64], C2[16 * 64];

    const int tid   = threadIdx.x;
    const int lane  = tid & 31;
    const int wv    = tid >> 5;
    const int bbase = blockIdx.x * 16;

    const v16h d1h0 = load_b_frag(W1h, 64, wv * 16,  0, lane);
    const v16h d1h1 = load_b_frag(W1h, 64, wv * 16, 32, lane);
    const v16h d2i0 = load_b_frag(W2i, 64, wv * 16,  0, lane);
    const v16h d2i1 = load_b_frag(W2i, 64, wv * 16, 32, lane);
    const v16h d2h0 = load_b_frag(W2h, 64, wv * 16,  0, lane);
    const v16h d2h1 = load_b_frag(W2h, 64, wv * 16, 32, lane);
    const float bv2 = bd2[wv * 16 + (lane & 15)];

    // Stage dec_in tile (f16) into cols [0,64); zero rest + state.
    for (int e = tid; e < 16 * 64; e += 512) {
        const int r = e >> 6, n = e & 63;
        Ab[r * 128 + n]      = (_Float16)dec_in[(size_t)(bbase + r) * HID + n];
        Ab[r * 128 + 64 + n] = (_Float16)0.0f;
        C1[e] = 0.0f; C2[e] = 0.0f;
    }
    __syncthreads();

    // pre1 = bd1 + dec_in @ d1_Wih^T -- constant across steps, lives in VGPRs.
    v8f pre1 = bias_acc(bd1[wv * 16 + (lane & 15)]);
    {
        const v16h di0 = load_b_frag(W1i, 64, wv * 16,  0, lane);
        const v16h di1 = load_b_frag(W1i, 64, wv * 16, 32, lane);
        const v16h a0  = load_a_frag(Ab, 128,  0, lane);
        const v16h a1  = load_a_frag(Ab, 128, 32, lane);
        pre1 = WMMA(a0, di0, pre1);
        pre1 = WMMA(a1, di1, pre1);
    }
    __syncthreads();

    // Reset cols [0,64) to zero h1 before the recurrence.
    for (int e = tid; e < 16 * 64; e += 512)
        Ab[(e >> 6) * 128 + (e & 63)] = (_Float16)0.0f;
    __syncthreads();

    for (int t = 0; t < TSTEPS; ++t) {
        // gates1 = pre1 + h1 @ d1_Whh^T   (2 WMMAs / wave)
        {
            const v16h a0 = load_a_frag(Ab, 128,  0, lane);
            const v16h a1 = load_a_frag(Ab, 128, 32, lane);
            v8f acc = pre1;
            acc = WMMA(a0, d1h0, acc);
            acc = WMMA(a1, d1h1, acc);
            store_c_tile(G, wv, lane, acc);
        }
        __syncthreads();

        cell_update(G, C1, Ab, 128, tid);              // h1 -> cols [0,64)
        cell_update(G, C1, Ab, 128, tid + 512);
        __syncthreads();

        // gates2 = bd2 + h1 @ d2_Wih^T + h2 @ d2_Whh^T   (4 WMMAs / wave)
        {
            const v16h a0 = load_a_frag(Ab, 128,  0, lane);
            const v16h a1 = load_a_frag(Ab, 128, 32, lane);
            const v16h a2 = load_a_frag(Ab, 128, 64, lane);
            const v16h a3 = load_a_frag(Ab, 128, 96, lane);
            v8f acc = bias_acc(bv2);
            acc = WMMA(a0, d2i0, acc);
            acc = WMMA(a1, d2i1, acc);
            acc = WMMA(a2, d2h0, acc);
            acc = WMMA(a3, d2h1, acc);
            store_c_tile(G, wv, lane, acc);
        }
        __syncthreads();

        // cell2 -> h2 (cols [64,128)); stream h2 to out[t] (write-once -> NT)
        {
            float* orow = out + (size_t)t * BATCH * HID + (size_t)bbase * HID;
            const float h_a = cell_update(G, C2, Ab + 64, 128, tid);
            __builtin_nontemporal_store(h_a, orow + tid);
            const float h_b = cell_update(G, C2, Ab + 64, 128, tid + 512);
            __builtin_nontemporal_store(h_b, orow + tid + 512);
        }
        __syncthreads();
    }
}

extern "C" void kernel_launch(void* const* d_in, const int* in_sizes, int n_in,
                              void* d_out, int out_size, void* d_ws, size_t ws_size,
                              hipStream_t stream) {
    const float* x     = (const float*)d_in[0];
    const float* Wih0  = (const float*)d_in[1];
    const float* Whh0  = (const float*)d_in[2];
    const float* b0    = (const float*)d_in[3];
    const float* Wih1  = (const float*)d_in[4];
    const float* Whh1  = (const float*)d_in[5];
    const float* b1    = (const float*)d_in[6];
    const float* d1Wih = (const float*)d_in[7];
    const float* d1Whh = (const float*)d_in[8];
    const float* d1b   = (const float*)d_in[9];
    const float* d2Wih = (const float*)d_in[10];
    const float* d2Whh = (const float*)d_in[11];
    const float* d2b   = (const float*)d_in[12];
    float* out    = (float*)d_out;
    float* dec_in = (float*)d_ws;   // B*64 f32 = 1 MB handoff buffer

    const int blocks = BATCH / 16;  // 256 independent batch tiles
    lstm_encoder<<<blocks, 512, 0, stream>>>(x, Wih0, Whh0, b0, Wih1, Whh1, b1,
                                             dec_in);
    lstm_decoder<<<blocks, 512, 0, stream>>>(dec_in, d1Wih, d1Whh, d1b,
                                             d2Wih, d2Whh, d2b, out);
}
